// MaxPoolingAggregator_35699768164471
// MI455X (gfx1250) — matrix operand order repaired and checked by reading, not compiled, not run on hardware
//
#include <hip/hip_runtime.h>

typedef __attribute__((ext_vector_type(16))) __bf16 v16bf;
typedef __attribute__((ext_vector_type(8)))  float  v8f;

// Problem constants (from reference)
#define NN     50000
#define KNBR   32
#define DIN    128
#define DHID   256
#define DOUT   128

// Packed-weight fragment counts (elements of bf16)
#define MLP_ELEMS ((DIN / 32) * (DHID / 16) * 512)   // 4*16*512 = 32768
#define NO_ELEMS  ((DHID / 32) * (DOUT / 16) * 512)  // 8*8*512  = 32768
#define IO_ELEMS  ((DIN / 32) * (DOUT / 16) * 512)   // 4*8*512  = 16384
#define PK_TOTAL  (MLP_ELEMS + NO_ELEMS + IO_ELEMS)  // 81920

__device__ __forceinline__ v8f wmma_bf16(v16bf a, v16bf b, v8f c) {
  // (neg_a, A, neg_b, B, c_mod, C, reuse_a, reuse_b)
  return __builtin_amdgcn_wmma_f32_16x16x32_bf16(false, a, false, b, (short)0, c,
                                                 false, false);
}

// Build a 16-element bf16 A-fragment chunk for one lane from two 8-float runs.
// A layout (16-bit A 16x32): lane (m = l&15, h = l>>4) holds
//   elems 0..7  = K = kc*32 + 8h   .. +7
//   elems 8..15 = K = kc*32 + 16 + 8h .. +7
__device__ __forceinline__ v16bf cvt16(const float* p0, const float* p1) {
  v16bf r;
#pragma unroll
  for (int i = 0; i < 8; ++i) {
    r[i]     = (__bf16)p0[i];
    r[i + 8] = (__bf16)p1[i];
  }
  return r;
}

__device__ __forceinline__ v16bf cvt16_sum(const float* p0, const float* p1,
                                           float& s) {
  v16bf r;
#pragma unroll
  for (int i = 0; i < 8; ++i) {
    float a = p0[i], b = p1[i];
    s += a + b;
    r[i]     = (__bf16)a;
    r[i + 8] = (__bf16)b;
  }
  return r;
}

// ---------------------------------------------------------------------------
// Prep kernel: pack W (f32 row-major [Krows x Ncols]) into bf16 B-fragments.
// B layout (16-bit B 32x16): lane l: n = l&15; element j (0..15) ->
// K = (l>>4)*8 + (j&7) + (j>>3)*16 (matches A fragment order).
// Fragment (kt, nt) stored contiguously: 32 lanes * 16 bf16 = 512 elems.
// ---------------------------------------------------------------------------
__device__ __forceinline__ void pack_one(const float* W, int Ncols, __bf16* dst,
                                         int idx) {
  int j    = idx & 15;
  int lane = (idx >> 4) & 31;
  int tile = idx >> 9;
  int ntiles = Ncols >> 4;
  int kt = tile / ntiles;
  int nt = tile - kt * ntiles;
  int k  = kt * 32 + (lane >> 4) * 8 + (j & 7) + (j >> 3) * 16;
  int n  = nt * 16 + (lane & 15);
  dst[idx] = (__bf16)W[k * Ncols + n];
}

__global__ void pack_weights_kernel(const float* __restrict__ Wmlp,
                                    const float* __restrict__ Wno,
                                    const float* __restrict__ Wio,
                                    void* __restrict__ ws) {
  __bf16* pk = (__bf16*)ws;
  int t = blockIdx.x * blockDim.x + threadIdx.x;
  if (t < MLP_ELEMS) {
    pack_one(Wmlp, DHID, pk, t);
  } else if (t < MLP_ELEMS + NO_ELEMS) {
    pack_one(Wno, DOUT, pk + MLP_ELEMS, t - MLP_ELEMS);
  } else if (t < PK_TOTAL) {
    pack_one(Wio, DOUT, pk + MLP_ELEMS + NO_ELEMS, t - (MLP_ELEMS + NO_ELEMS));
  }
}

// ---------------------------------------------------------------------------
// Main kernel: 256 threads = 8 waves; block handles 16 nodes (50000 = 16*3125).
// Phase 0: stage packed W_mlp fragments (64KB bf16) + b_mlp into LDS.
// Phase 1: each wave does 2 nodes: [32x128] @ W_mlp -> relu/mask -> maxpool,
//          result rows staged in LDS. B comes from LDS (ds_load, short dscnt
//          waits) so the vmem pipe is dedicated to the 819MB neighbor stream.
// Phase 2: wave w computes output N-tiles w (self: x@W_io) and 8+w
//          (neigh: hmax@W_no) for the 16-node batch, relu, store.
// ---------------------------------------------------------------------------
__global__ __launch_bounds__(256) void
maxpool_agg_kernel(const float* __restrict__ x, const float* __restrict__ nbr,
                   const float* __restrict__ b_mlp,
                   const float* __restrict__ b_no,
                   const float* __restrict__ b_io, const void* __restrict__ ws,
                   float* __restrict__ out) {
  const __bf16* wmlp_g = (const __bf16*)ws;
  const __bf16* wno    = wmlp_g + MLP_ELEMS;
  const __bf16* wio    = wno + NO_ELEMS;

  const int tid  = threadIdx.x;
  const int lane = tid & 31;
  const int wave = tid >> 5;
  const int l15  = lane & 15;
  const int lh   = lane >> 4;
  const int nodeBase = blockIdx.x * 16;

  // LDS: staged W_mlp fragments + b_mlp + hmax (stride 260 => 16 rows at the
  // same float4 column offset land on 16 disjoint bank groups).
  __shared__ __align__(16) __bf16 wmlp_s[MLP_ELEMS];  // 64 KB
  __shared__ float bmlp_s[DHID];                      // 1 KB
  __shared__ float hmax_lds[16][260];                 // 16.6 KB

  // ---------------- Phase 0: stage weights into LDS ------------------------
  {
    const uint4* s = (const uint4*)wmlp_g;  // 4096 x 16B
    uint4* d = (uint4*)wmlp_s;
#pragma unroll
    for (int i = 0; i < (MLP_ELEMS / 8) / 256; ++i)
      d[tid + i * 256] = s[tid + i * 256];
    if (tid < DHID) bmlp_s[tid] = b_mlp[tid];
  }
  __syncthreads();

  // ---------------- Phase 1: neighbor MLP + masked max-pool ----------------
#pragma unroll
  for (int sub = 0; sub < 2; ++sub) {
    const int ln   = 2 * wave + sub;
    const int node = nodeBase + ln;
    const float* nbp = nbr + (size_t)node * (KNBR * DIN);

    // prefetch next node's 16KB neighbor tile: 4 x 32 lanes x 128B lines
    {
      const char* nxt = (const char*)(nbp + KNBR * DIN) + lane * 512;
#pragma unroll
      for (int i = 0; i < 4; ++i) __builtin_prefetch(nxt + i * 128, 0, 0);
    }

    v16bf A[2][4];
    unsigned rowmask[2];
#pragma unroll
    for (int mt = 0; mt < 2; ++mt) {
      const float* rp = nbp + (size_t)(mt * 16 + l15) * DIN;
      float rs = 0.0f;
#pragma unroll
      for (int kc = 0; kc < 4; ++kc) {
        const int b0 = kc * 32 + lh * 8;
        A[mt][kc] = cvt16_sum(rp + b0, rp + b0 + 16, rs);
      }
      // full f32 row sum lives split across lanes l and l^16
      const float full = rs + __shfl_xor(rs, 16, 32);
      rowmask[mt] = (unsigned)__ballot(full == 0.0f);
    }

    for (int nt = 0; nt < 16; ++nt) {
      v8f a0 = {};
      v8f a1 = {};
#pragma unroll
      for (int kc = 0; kc < 4; ++kc) {
        v16bf B = *(const v16bf*)(wmlp_s + (kc * 16 + nt) * 512 + lane * 16);
        a0 = wmma_bf16(A[0][kc], B, a0);
        a1 = wmma_bf16(A[1][kc], B, a1);
      }
      const float bias = bmlp_s[nt * 16 + l15];
      float cm = 0.0f;  // relu floor: all-masked rows pool to 0, matching ref
#pragma unroll
      for (int j = 0; j < 8; ++j) {
        const int m = lh * 8 + j;  // row within 16-row tile for this element
        float h0 = fmaxf(a0[j] + bias, 0.0f);
        if ((rowmask[0] >> m) & 1u) h0 = 0.0f;
        float h1 = fmaxf(a1[j] + bias, 0.0f);
        if ((rowmask[1] >> m) & 1u) h1 = 0.0f;
        cm = fmaxf(cm, fmaxf(h0, h1));
      }
      cm = fmaxf(cm, __shfl_xor(cm, 16, 32));  // combine row halves
      if (lh == 0) hmax_lds[ln][nt * 16 + l15] = cm;
    }
  }

  __syncthreads();

  // ---------------- Phase 2: self_out and neigh_out via WMMA ---------------
  // Self tile: [16 nodes x 128] @ W_io -> cols [wave*16, wave*16+16)
  {
    v8f acc = {};
#pragma unroll
    for (int kc = 0; kc < 4; ++kc) {
      const float* rp = x + (size_t)(nodeBase + l15) * DIN + kc * 32 + lh * 8;
      v16bf Af = cvt16(rp, rp + 16);
      v16bf B  = *(const v16bf*)(wio + (size_t)((kc * 8 + wave) * 512 + lane * 16));
      acc = wmma_bf16(Af, B, acc);
    }
    const float bias = b_io[wave * 16 + l15];
#pragma unroll
    for (int j = 0; j < 8; ++j) {
      const int m = lh * 8 + j;
      out[(size_t)(nodeBase + m) * (2 * DOUT) + wave * 16 + l15] =
          fmaxf(acc[j] + bias, 0.0f);
    }
  }
  // Neighbor tile: [16 nodes x 256] @ W_no -> cols 128 + [wave*16, ...)
  {
    v8f acc = {};
#pragma unroll
    for (int kc = 0; kc < 8; ++kc) {
      const float* rp = &hmax_lds[l15][kc * 32 + lh * 8];
      v16bf Af = cvt16(rp, rp + 16);
      v16bf B  = *(const v16bf*)(wno + (size_t)((kc * 8 + wave) * 512 + lane * 16));
      acc = wmma_bf16(Af, B, acc);
    }
    const float bias = b_no[wave * 16 + l15];
#pragma unroll
    for (int j = 0; j < 8; ++j) {
      const int m = lh * 8 + j;
      out[(size_t)(nodeBase + m) * (2 * DOUT) + DOUT + wave * 16 + l15] =
          fmaxf(acc[j] + bias, 0.0f);
    }
  }
}

// ---------------------------------------------------------------------------
extern "C" void kernel_launch(void* const* d_in, const int* in_sizes, int n_in,
                              void* d_out, int out_size, void* d_ws,
                              size_t ws_size, hipStream_t stream) {
  const float* x     = (const float*)d_in[0];
  const float* nbr   = (const float*)d_in[1];
  const float* Wmlp  = (const float*)d_in[2];
  const float* bmlp  = (const float*)d_in[3];
  const float* Wno   = (const float*)d_in[4];
  const float* bno   = (const float*)d_in[5];
  const float* Wio   = (const float*)d_in[6];
  const float* bio   = (const float*)d_in[7];
  float* out = (float*)d_out;

  (void)in_sizes; (void)n_in; (void)out_size; (void)ws_size;

  // Pack all three weight matrices into bf16 WMMA B-fragments in d_ws.
  pack_weights_kernel<<<(PK_TOTAL + 255) / 256, 256, 0, stream>>>(Wmlp, Wno,
                                                                  Wio, d_ws);

  // 16 nodes per block: 50000 / 16 = 3125 exactly.
  maxpool_agg_kernel<<<NN / 16, 256, 0, stream>>>(x, nbr, bmlp, bno, bio, d_ws,
                                                  out);
}
